// GCNEncoder_35261681500771
// MI455X (gfx1250) — matrix operand order repaired and checked by reading, not compiled, not run on hardware
//
#include <hip/hip_runtime.h>
#include <stddef.h>

// ---------------------------------------------------------------------------
// GCN 2-layer forward for MI455X (gfx1250, wave32, WMMA)
//   layer: t = X @ W (f16 WMMA, f32 acc) ; agg = scatter_add(t[src]*norm, dst)
//          + selfloop (dinv^2) + bias ; relu (layer 1 only)
// ---------------------------------------------------------------------------

#define N_NODES 50000
#define N_EDGES 600000
#define IN_CH   128
#define HID     128
#define OUT_CH  64

typedef __attribute__((ext_vector_type(16))) _Float16 v16h;
typedef __attribute__((ext_vector_type(8)))  _Float16 v8h;
typedef __attribute__((ext_vector_type(8)))  float    v8f;

union V16H { v16h v; v8h h[2]; };

// LDS row pitch in halfs: 136 halfs = 272B = 68 banks -> 4-bank rotation per
// row -> conflict-free 16B fragment loads across 16 consecutive rows.
#define PITCH 136

// ---------------------------------------------------------------------------
// WMMA GEMM: Y[nrows, NCOLS] = X[nrows, 128] @ W[128, NCOLS]
// Block = 256 threads (8 waves), owns 128 rows. Wave w owns M-tile w and all
// NCOLS/16 N-tiles. K loop = 4 steps of 32 (v_wmma_f32_16x16x32_f16).
// ---------------------------------------------------------------------------
template <int NCOLS>
__global__ __launch_bounds__(256) void gemm_wmma(const float* __restrict__ X,
                                                 const float* __restrict__ W,
                                                 float* __restrict__ Y,
                                                 int nrows) {
    __shared__ __align__(16) _Float16 sA[128 * PITCH];     // X tile, f16
    __shared__ __align__(16) _Float16 sB[NCOLS * PITCH];   // W transposed, f16

    const int tid = threadIdx.x;
    const int rowBase = blockIdx.x * 128;

    // Stage X rows [rowBase, rowBase+128) as f16 (float4 coalesced loads).
    for (int i = tid; i < 128 * 128 / 4; i += 256) {
        const int l = i * 4;
        const int r = l >> 7;
        const int k = l & 127;
        float4 v = {0.f, 0.f, 0.f, 0.f};
        if (rowBase + r < nrows)
            v = *(const float4*)(X + (size_t)(rowBase + r) * 128 + k);
        _Float16* p = &sA[r * PITCH + k];
        p[0] = (_Float16)v.x; p[1] = (_Float16)v.y;
        p[2] = (_Float16)v.z; p[3] = (_Float16)v.w;
    }
    // Stage W transposed: sB[n*PITCH + k] = W[k*NCOLS + n] (coalesced reads).
    for (int i = tid; i < 128 * NCOLS; i += 256) {
        const int k = i / NCOLS;
        const int n = i % NCOLS;
        sB[n * PITCH + k] = (_Float16)W[i];
    }
    __syncthreads();

    const int wave   = tid >> 5;
    const int lane   = tid & 31;
    const int laneLo = lane & 15;
    const int hiHalf = lane >> 4;        // 0: lanes 0-15, 1: lanes 16-31
    const int aRow   = wave * 16 + laneLo;

    constexpr int NT = NCOLS / 16;
    v8f acc[NT];
#pragma unroll
    for (int t = 0; t < NT; ++t)
#pragma unroll
        for (int j = 0; j < 8; ++j) acc[t][j] = 0.0f;

#pragma unroll
    for (int kstep = 0; kstep < 4; ++kstep) {
        const int k0 = kstep * 32;
        // A fragment (ISA 16-bit A 16x32 layout):
        //  elems 0..7  -> K = k0 + hiHalf*8 + {0..7}
        //  elems 8..15 -> K = k0 + 16 + hiHalf*8 + {0..7}
        V16H a;
        a.h[0] = *(const v8h*)&sA[aRow * PITCH + k0 + hiHalf * 8];
        a.h[1] = *(const v8h*)&sA[aRow * PITCH + k0 + 16 + hiHalf * 8];
#pragma unroll
        for (int t = 0; t < NT; ++t) {
            // B fragment: col = t*16 + laneLo, K = k0 + hiHalf*8 + {0..7}, k0 + 16 + hiHalf*8 + {0..7}
            const _Float16* bp = &sB[(t * 16 + laneLo) * PITCH + k0 + hiHalf * 8];
            V16H b;
            b.h[0] = *(const v8h*)(bp);
            b.h[1] = *(const v8h*)(bp + 16);
            acc[t] = __builtin_amdgcn_wmma_f32_16x16x32_f16(
                false, a.v, false, b.v, (short)0, acc[t], false, false);
        }
    }

    // D layout: VGPR v -> row = hiHalf*8 + v, col = laneLo (per 16x16 tile).
#pragma unroll
    for (int t = 0; t < NT; ++t) {
#pragma unroll
        for (int v = 0; v < 8; ++v) {
            const int r = rowBase + wave * 16 + hiHalf * 8 + v;
            if (r < nrows)
                Y[(size_t)r * NCOLS + t * 16 + laneLo] = acc[t][v];
        }
    }
}

// ---------------------------------------------------------------------------
// Degree / normalization kernels
// ---------------------------------------------------------------------------
__global__ void k_deg_init(float* deg) {
    const int i = blockIdx.x * blockDim.x + threadIdx.x;
    if (i < N_NODES) deg[i] = 1.0f;   // self-loop contributes 1
}

__global__ void k_deg_accum(const long long* __restrict__ dst, float* deg) {
    const int e = blockIdx.x * blockDim.x + threadIdx.x;
    if (e < N_EDGES) atomicAdd(&deg[(int)dst[e]], 1.0f);
}

__global__ void k_dinv(float* deg) {
    const int i = blockIdx.x * blockDim.x + threadIdx.x;
    if (i < N_NODES) deg[i] = rsqrtf(deg[i]);   // deg >= 1 (self-loop)
}

// agg = t * dinv^2 (self-loop term) + bias
template <int C>
__global__ void k_init_agg(const float* __restrict__ t,
                           const float* __restrict__ dinv,
                           const float* __restrict__ bias,
                           float* __restrict__ agg) {
    const int idx = blockIdx.x * blockDim.x + threadIdx.x;
    if (idx >= N_NODES * C) return;
    constexpr int SH = (C == 128) ? 7 : 6;
    const int r = idx >> SH;
    const int c = idx & (C - 1);
    const float di = dinv[r];
    agg[idx] = t[idx] * di * di + bias[c];
}

// One wave per edge; lanes stride channels -> coalesced gather + atomics.
template <int C>
__global__ __launch_bounds__(256) void k_scatter(const long long* __restrict__ ei,
                                                 const float* __restrict__ dinv,
                                                 const float* __restrict__ t,
                                                 float* __restrict__ agg) {
    const int gw   = (blockIdx.x * 256 + threadIdx.x) >> 5;  // wave-uniform
    const int lane = threadIdx.x & 31;
    if (gw >= N_EDGES) return;
    const int s = (int)ei[gw];
    const int d = (int)ei[N_EDGES + gw];
    const float w = dinv[s] * dinv[d];
    const float* ts = t + (size_t)s * C;
    float* ad = agg + (size_t)d * C;
#pragma unroll
    for (int c = lane; c < C; c += 32)
        atomicAdd(ad + c, ts[c] * w);
}

__global__ void k_relu(float* __restrict__ a, int n) {
    const int i = blockIdx.x * blockDim.x + threadIdx.x;
    if (i < n) a[i] = fmaxf(a[i], 0.0f);
}

// ---------------------------------------------------------------------------
// Launch
// ---------------------------------------------------------------------------
extern "C" void kernel_launch(void* const* d_in, const int* in_sizes, int n_in,
                              void* d_out, int out_size, void* d_ws, size_t ws_size,
                              hipStream_t stream) {
    const float*     x  = (const float*)d_in[0];
    const long long* ei = (const long long*)d_in[1];   // int64 [2, E]
    const float*     W1 = (const float*)d_in[2];
    const float*     b1 = (const float*)d_in[3];
    const float*     W2 = (const float*)d_in[4];
    const float*     b2 = (const float*)d_in[5];
    float*           out = (float*)d_out;

    // Workspace layout (floats): deg | t1[N*HID] | a1[N*HID] | t2[N*OUT]
    float* deg = (float*)d_ws;
    float* t1  = deg + N_NODES;
    float* a1  = t1 + (size_t)N_NODES * HID;
    float* t2  = a1 + (size_t)N_NODES * HID;

    const dim3 B(256);
    const int gN   = (N_NODES + 255) / 256;
    const int gE   = (N_EDGES + 255) / 256;
    const int gEw  = (N_EDGES * 32 + 255) / 256;       // wave-per-edge
    const int gRow = (N_NODES + 127) / 128;            // GEMM blocks
    const int gH   = (N_NODES * HID + 255) / 256;
    const int gO   = (N_NODES * OUT_CH + 255) / 256;

    // Normalization: deg (with self-loops) -> dinv in place.
    k_deg_init<<<gN, B, 0, stream>>>(deg);
    k_deg_accum<<<gE, B, 0, stream>>>(ei + N_EDGES, deg);
    k_dinv<<<gN, B, 0, stream>>>(deg);

    // Layer 1: t1 = x@W1 ; a1 = scatter + selfloop + b1 ; relu
    gemm_wmma<HID><<<gRow, B, 0, stream>>>(x, W1, t1, N_NODES);
    k_init_agg<HID><<<gH, B, 0, stream>>>(t1, deg, b1, a1);
    k_scatter<HID><<<gEw, B, 0, stream>>>(ei, deg, t1, a1);
    k_relu<<<gH, B, 0, stream>>>(a1, N_NODES * HID);

    // Layer 2: t2 = a1@W2 ; out = scatter + selfloop + b2
    gemm_wmma<OUT_CH><<<gRow, B, 0, stream>>>(a1, W2, t2, N_NODES);
    k_init_agg<OUT_CH><<<gO, B, 0, stream>>>(t2, deg, b2, out);
    k_scatter<OUT_CH><<<gEw, B, 0, stream>>>(ei, deg, t2, out);
}
